// LeakyIntegrateFireAttention_3384434229249
// MI455X (gfx1250) — matrix operand order repaired and checked by reading, not compiled, not run on hardware
//
#include <hip/hip_runtime.h>
#include <hip/hip_bf16.h>
#include <math.h>

typedef __attribute__((ext_vector_type(16))) _Float16     v16h;
typedef __attribute__((ext_vector_type(8)))  float        v8f;
typedef __attribute__((ext_vector_type(4)))  unsigned int u32x4;
typedef __attribute__((ext_vector_type(8)))  unsigned int u32x8;

#define BB 2
#define TT 2048u
#define CC 1024u
#define HH 16u
#define HD 64u
#define M_ROWS (BB*TT)   // 4096
#define NQKV   (3u*CC)   // 3072

union VPack { u32x4 u[2]; v16h v; };

// ---- WMMA operand loaders (wave32, v_wmma_f32_16x16x32_f16) ----
// A-matrix 16x32 (MxK), row-major source, contiguous k order used here (k = 16*half + s):
//   lanes 0-15 : row=lane,    K = 0..15
//   lanes 16-31: row=lane-16, K = 16..31
__device__ __forceinline__ v16h load_a16(const _Float16* base, unsigned ld) {
    unsigned lane = threadIdx.x & 31u;
    unsigned half = lane >> 4;
    unsigned m    = lane & 15u;
    const u32x4* p = (const u32x4*)(base + m * ld);
    VPack t;
    t.u[0] = p[2u * half];        // halves [half*16 .. half*16+7]
    t.u[1] = p[2u * half + 1u];   // halves [half*16+8 .. half*16+15]
    return t.v;
}

// B-matrix 32x16 (KxN): column n contiguous in memory (leading dim = ld
// between columns). Lanes 0-15 hold K=0..15 of col=lane, lanes 16-31 hold
// K=16..31 of col=lane-16. One aligned 32B run per lane.
__device__ __forceinline__ v16h load_b16(const _Float16* base, unsigned ld) {
    unsigned lane = threadIdx.x & 31u;
    unsigned half = lane >> 4;
    unsigned n    = lane & 15u;
    const u32x4* p = (const u32x4*)(base + n * ld + half * 16u);
    VPack t;
    t.u[0] = p[0];
    t.u[1] = p[1];
    return t.v;
}

__device__ __forceinline__ v8f wmma_f16(v16h a, v16h b, v8f c) {
    return __builtin_amdgcn_wmma_f32_16x16x32_f16(false, a, false, b,
                                                  (short)0, c, false, false);
}

// ---- Tensor Data Mover: DMA one 16x64 f16 row-major tile into LDS ----
// D# group0: count=1, lds_addr[63:32], global_addr[120:64], type=2[127:126]
// D# group1: data_size=2B, tensor_dim0=64, tensor_dim1=16, tile_dim0=64,
//            tile_dim1=16, tensor_dim0_stride=64 (2D tile, no padding).
__device__ __forceinline__ void tdm_load_k_tile(const _Float16* gaddr,
                                                unsigned lds_off) {
    unsigned long long ga = (unsigned long long)(size_t)gaddr;
    u32x4 g0;
    g0.x = 1u;                                  // count=1, user descriptor
    g0.y = lds_off;                             // lds_addr (bytes)
    g0.z = (unsigned)ga;                        // global_addr[31:0]
    g0.w = (unsigned)(ga >> 32) | (2u << 30);   // global_addr[56:32] | type=2
    u32x8 g1;
    g1[0] = 1u << 16;                           // data_size = 2 bytes
    g1[1] = HD << 16;                           // tensor_dim0 = 64
    g1[2] = 16u << 16;                          // tensor_dim1 = 16
    g1[3] = HD << 16;                           // tile_dim0 = 64
    g1[4] = 16u;                                // tile_dim1 = 16
    g1[5] = HD;                                 // tensor_dim0_stride = 64
    g1[6] = 0u;
    g1[7] = 0u;
    asm volatile("tensor_load_to_lds %0, %1" :: "s"(g0), "s"(g1) : "memory");
}

// ---- conversion kernels ----
__global__ void cvt_f32_to_f16_kernel(const float* __restrict__ src,
                                      _Float16* __restrict__ dst, unsigned n) {
    unsigned i = blockIdx.x * blockDim.x + threadIdx.x;
    if (i < n) dst[i] = (_Float16)src[i];
}

// src [rows,cols] row-major f32 -> dst [cols,rows] f16 (transposed)
__global__ void transpose_f32_to_f16_kernel(const float* __restrict__ src,
                                            _Float16* __restrict__ dst,
                                            unsigned rows, unsigned cols) {
    unsigned i = blockIdx.x * blockDim.x + threadIdx.x;
    if (i < rows * cols) {
        unsigned r = i / cols, c = i % cols;
        dst[(size_t)c * rows + r] = (_Float16)src[i];
    }
}

// ---- QKV GEMM: [4096,1024] x [1024,3072] + bias; scatter into Q/K/Vt ----
// Each wave computes a 32x64 block: 2 A-tiles x 4 B-tiles, 8 accumulators.
__global__ void qkv_gemm_kernel(const _Float16* __restrict__ Xh,
                                const _Float16* __restrict__ WhT,
                                const float* __restrict__ bias,
                                _Float16* __restrict__ Qh,
                                _Float16* __restrict__ Kh,
                                _Float16* __restrict__ VhT) {
    unsigned wave = threadIdx.x >> 5;
    unsigned tile = blockIdx.x * (blockDim.x >> 5) + wave;
    const unsigned ntn = NQKV / 64u;        // 48 column blocks of 64
    unsigned mt = tile / ntn;
    unsigned nt = tile % ntn;

    const _Float16* Ab = Xh  + (size_t)(mt * 32u) * CC;
    const _Float16* Bb = WhT + (size_t)(nt * 64u) * CC;

    v8f acc[2][4];
#pragma unroll
    for (int i = 0; i < 2; ++i)
#pragma unroll
        for (int j = 0; j < 4; ++j) acc[i][j] = (v8f){};

    for (unsigned k0 = 0; k0 < CC; k0 += 32u) {
        __builtin_prefetch(Ab + k0 + 128u, 0, 3);
        __builtin_prefetch(Bb + k0 + 128u, 0, 3);
        v16h a0 = load_a16(Ab + k0,            CC);
        v16h a1 = load_a16(Ab + 16u * CC + k0, CC);
#pragma unroll
        for (unsigned j = 0; j < 4; ++j) {
            v16h b = load_b16(Bb + (size_t)(j * 16u) * CC + k0, CC);
            acc[0][j] = wmma_f16(a0, b, acc[0][j]);
            acc[1][j] = wmma_f16(a1, b, acc[1][j]);
        }
    }

    unsigned lane = threadIdx.x & 31u;
    unsigned half = lane >> 4;
    unsigned n    = lane & 15u;
#pragma unroll
    for (unsigned j = 0; j < 4; ++j) {
        unsigned col   = nt * 64u + j * 16u + n;
        unsigned which = col / CC;          // 0=q 1=k 2=v  (shift)
        unsigned rem   = col & (CC - 1u);
        unsigned h = rem / HD;
        unsigned d = rem & (HD - 1u);
        float bv = bias[col];
#pragma unroll
        for (unsigned i = 0; i < 2; ++i) {
#pragma unroll
            for (unsigned r = 0; r < 8; ++r) {
                unsigned row  = mt * 32u + i * 16u + r + half * 8u;
                unsigned bidx = row / TT;
                unsigned t    = row & (TT - 1u);
                unsigned bh   = bidx * HH + h;
                float v = acc[i][j][r] + bv;
                if (which == 0u)      Qh[((size_t)bh * TT + t) * HD + d]  = (_Float16)v;
                else if (which == 1u) Kh[((size_t)bh * TT + t) * HD + d]  = (_Float16)v;
                else                  VhT[((size_t)bh * HD + d) * TT + t] = (_Float16)v;
            }
        }
    }
}

// ---- LIF attention: one wave per (b,h,q-tile of 16 rows) ----
__global__ void lif_attention_kernel(const _Float16* __restrict__ Qh,
                                     const _Float16* __restrict__ Kh,
                                     const _Float16* __restrict__ VhT,
                                     const float* __restrict__ threshold,
                                     const float* __restrict__ leak,
                                     const float* __restrict__ steep,
                                     _Float16* __restrict__ Yh) {
    __shared__ _Float16 smod[16][32];        // mod tile C->A layout staging
    __shared__ _Float16 kbuf[2][16 * 64];    // TDM double buffer (2 x 2KB)

    const unsigned QT = TT / 16u;            // 128
    unsigned qt = blockIdx.x & (QT - 1u);
    unsigned bh = blockIdx.x / QT;           // 0..B*H-1
    unsigned h  = bh & (HH - 1u);

    unsigned lane = threadIdx.x & 31u;
    unsigned half = lane >> 4;
    unsigned n    = lane & 15u;

    const _Float16* Qbase = Qh + ((size_t)bh * TT + qt * 16u) * HD;
    v16h aq0 = load_a16(Qbase,       HD);    // d = 0..31
    v16h aq1 = load_a16(Qbase + 32u, HD);    // d = 32..63

    const float scale = 0.125f;              // 1/sqrt(64)
    float thr = fabsf(threshold[h]) * 0.1f;
    float lk  = 1.0f / (1.0f + __expf(-leak[h]));
    float st  = logf(1.0f + __expf(steep[h]));

    float m[8], l[8], msum[8];
#pragma unroll
    for (int r = 0; r < 8; ++r) { m[r] = -1e30f; l[r] = 0.0f; msum[r] = 0.0f; }

    const _Float16* Kall = Kh + (size_t)bh * TT * HD;
    // LDS byte offsets (low 32 bits of the generic/shared address)
    unsigned koff0 = (unsigned)(size_t)&kbuf[0][0];
    unsigned koff1 = (unsigned)(size_t)&kbuf[1][0];

    // ---- pass 1: online row max + sum of exp; K tiles streamed by TDM ----
    tdm_load_k_tile(Kall, koff0);
    for (unsigned kt = 0; kt <= qt; ++kt) {
        unsigned cur = kt & 1u;
        if (kt < qt) {
            // prefetch next tile into the other buffer, then wait for current
            tdm_load_k_tile(Kall + (size_t)(kt + 1u) * 16u * HD,
                            cur ? koff0 : koff1);
            __builtin_amdgcn_s_wait_tensorcnt(1);
        } else {
            __builtin_amdgcn_s_wait_tensorcnt(0);
        }
        const _Float16* Kb = &kbuf[cur][0];
        v16h bk0 = load_b16(Kb,       HD);
        v16h bk1 = load_b16(Kb + 32u, HD);
        v8f c = {};
        c = wmma_f16(aq0, bk0, c);
        c = wmma_f16(aq1, bk1, c);
        unsigned kcol = kt * 16u + n;
#pragma unroll
        for (unsigned r = 0; r < 8; ++r) {
            unsigned qrow = qt * 16u + r + half * 8u;
            float s  = (kcol <= qrow) ? c[r] * scale : -1e30f;
            float mo = m[r];
            float mn = fmaxf(mo, s);
            l[r] = l[r] * __expf(mo - mn) + __expf(s - mn);
            m[r] = mn;
        }
    }
    // cross-lane combine within each 16-lane half (rows r..r+7 / r+8..r+15)
#pragma unroll
    for (int mask = 1; mask <= 8; mask <<= 1) {
#pragma unroll
        for (int r = 0; r < 8; ++r) {
            float mo = __shfl_xor(m[r], mask, 32);
            float lo = __shfl_xor(l[r], mask, 32);
            float mn = fmaxf(m[r], mo);
            l[r] = l[r] * __expf(m[r] - mn) + lo * __expf(mo - mn);
            m[r] = mn;
        }
    }

    v8f cy0 = {}, cy1 = {}, cy2 = {}, cy3 = {};

    // ---- pass 2: probs -> LIF mod -> modsum + unnormalized y (WMMA) ----
    for (unsigned kt2 = 0; kt2 <= qt; kt2 += 2u) {
#pragma unroll
        for (unsigned sub = 0; sub < 2; ++sub) {
            unsigned kt = kt2 + sub;
            if (kt <= qt) {
                const _Float16* Kbase = Kall + (size_t)kt * 16u * HD;
                v16h bk0 = load_b16(Kbase,       HD);
                v16h bk1 = load_b16(Kbase + 32u, HD);
                v8f c = {};
                c = wmma_f16(aq0, bk0, c);
                c = wmma_f16(aq1, bk1, c);
                unsigned kcol = kt * 16u + n;
#pragma unroll
                for (unsigned r = 0; r < 8; ++r) {
                    unsigned row  = r + half * 8u;
                    unsigned qrow = qt * 16u + row;
                    float p = 0.0f;
                    if (kcol <= qrow) p = __expf(c[r] * scale - m[r]) / l[r];
                    float fire = 1.0f / (1.0f + __expf(-st * (p - thr)));
                    float w    = fire + lk * (1.0f - fire);
                    float mod  = p * w;
                    msum[r] += mod;
                    smod[row][sub * 16u + n] = (_Float16)mod;
                }
            } else {
#pragma unroll
                for (unsigned r = 0; r < 8; ++r)
                    smod[r + half * 8u][sub * 16u + n] = (_Float16)0.0f;
            }
        }
        __syncthreads();
        // re-read mod tile in A layout: two 16B runs -> 2x ds_load_b128
        v16h am;
        {
            unsigned mr = lane & 15u;
            const u32x4* p = (const u32x4*)&smod[mr][0];
            VPack t;
            t.u[0] = p[2u * half];
            t.u[1] = p[2u * half + 1u];
            am = t.v;
        }
        __syncthreads();

        unsigned t0 = kt2 * 16u;
        const _Float16* Vb = VhT + (size_t)bh * HD * TT + t0;
        v16h bv0 = load_b16(Vb + (size_t) 0u * TT, TT);
        v16h bv1 = load_b16(Vb + (size_t)16u * TT, TT);
        v16h bv2 = load_b16(Vb + (size_t)32u * TT, TT);
        v16h bv3 = load_b16(Vb + (size_t)48u * TT, TT);
        cy0 = wmma_f16(am, bv0, cy0);
        cy1 = wmma_f16(am, bv1, cy1);
        cy2 = wmma_f16(am, bv2, cy2);
        cy3 = wmma_f16(am, bv3, cy3);
    }

    // reduce modsum per row across the 16-lane half
#pragma unroll
    for (int mask = 1; mask <= 8; mask <<= 1) {
#pragma unroll
        for (int r = 0; r < 8; ++r)
            msum[r] += __shfl_xor(msum[r], mask, 32);
    }

    unsigned bidx = bh / HH;
#pragma unroll
    for (unsigned r = 0; r < 8; ++r) {
        unsigned row = r + half * 8u;
        unsigned t   = qt * 16u + row;
        float inv = 1.0f / (msum[r] + 1e-8f);
        _Float16* yp = Yh + ((size_t)bidx * TT + t) * CC + h * HD;
        yp[ 0u + n] = (_Float16)(cy0[r] * inv);
        yp[16u + n] = (_Float16)(cy1[r] * inv);
        yp[32u + n] = (_Float16)(cy2[r] * inv);
        yp[48u + n] = (_Float16)(cy3[r] * inv);
    }
}

// ---- output projection: [4096,1024] x [1024,1024] + bias -> f32 ----
__global__ void proj_gemm_kernel(const _Float16* __restrict__ Yh,
                                 const _Float16* __restrict__ WpT,
                                 const float* __restrict__ bias,
                                 float* __restrict__ out) {
    unsigned wave = threadIdx.x >> 5;
    unsigned tile = blockIdx.x * (blockDim.x >> 5) + wave;
    const unsigned ntn = CC / 64u;          // 16 column blocks
    unsigned mt = tile / ntn;
    unsigned nt = tile % ntn;

    const _Float16* Ab = Yh  + (size_t)(mt * 32u) * CC;
    const _Float16* Bb = WpT + (size_t)(nt * 64u) * CC;

    v8f acc[2][4];
#pragma unroll
    for (int i = 0; i < 2; ++i)
#pragma unroll
        for (int j = 0; j < 4; ++j) acc[i][j] = (v8f){};

    for (unsigned k0 = 0; k0 < CC; k0 += 32u) {
        __builtin_prefetch(Ab + k0 + 128u, 0, 3);
        __builtin_prefetch(Bb + k0 + 128u, 0, 3);
        v16h a0 = load_a16(Ab + k0,            CC);
        v16h a1 = load_a16(Ab + 16u * CC + k0, CC);
#pragma unroll
        for (unsigned j = 0; j < 4; ++j) {
            v16h b = load_b16(Bb + (size_t)(j * 16u) * CC + k0, CC);
            acc[0][j] = wmma_f16(a0, b, acc[0][j]);
            acc[1][j] = wmma_f16(a1, b, acc[1][j]);
        }
    }

    unsigned lane = threadIdx.x & 31u;
    unsigned half = lane >> 4;
    unsigned n    = lane & 15u;
#pragma unroll
    for (unsigned j = 0; j < 4; ++j) {
        unsigned col = nt * 64u + j * 16u + n;
        float bv = bias[col];
#pragma unroll
        for (unsigned i = 0; i < 2; ++i) {
#pragma unroll
            for (unsigned r = 0; r < 8; ++r) {
                unsigned row = mt * 32u + i * 16u + r + half * 8u;
                out[(size_t)row * CC + col] = acc[i][j][r] + bv;
            }
        }
    }
}

extern "C" void kernel_launch(void* const* d_in, const int* in_sizes, int n_in,
                              void* d_out, int out_size, void* d_ws, size_t ws_size,
                              hipStream_t stream) {
    (void)in_sizes; (void)n_in; (void)out_size; (void)ws_size;
    const float* x         = (const float*)d_in[0];
    const float* W_attn    = (const float*)d_in[1];
    const float* b_attn    = (const float*)d_in[2];
    const float* W_proj    = (const float*)d_in[3];
    const float* b_proj    = (const float*)d_in[4];
    const float* threshold = (const float*)d_in[5];
    const float* leak      = (const float*)d_in[6];
    const float* steep     = (const float*)d_in[7];
    float* out = (float*)d_out;

    _Float16* ws = (_Float16*)d_ws;
    size_t o = 0;
    _Float16* Xh  = ws + o; o += (size_t)M_ROWS * CC;       // 4M halves
    _Float16* WhT = ws + o; o += (size_t)NQKV * CC;         // 3M
    _Float16* WpT = ws + o; o += (size_t)CC * CC;           // 1M
    _Float16* Qh  = ws + o; o += (size_t)BB * HH * TT * HD; // 4M
    _Float16* Kh  = ws + o; o += (size_t)BB * HH * TT * HD; // 4M
    _Float16* VhT = ws + o; o += (size_t)BB * HH * TT * HD; // 4M
    _Float16* Yh  = ws + o;                                 // 4M

    unsigned nX = M_ROWS * CC;
    cvt_f32_to_f16_kernel<<<(nX + 255) / 256, 256, 0, stream>>>(x, Xh, nX);
    transpose_f32_to_f16_kernel<<<(CC * NQKV + 255) / 256, 256, 0, stream>>>(
        W_attn, WhT, CC, NQKV);
    transpose_f32_to_f16_kernel<<<(CC * CC + 255) / 256, 256, 0, stream>>>(
        W_proj, WpT, CC, CC);

    // (4096/32) x (3072/64) = 6144 waves, 8 waves/block -> 768 blocks
    qkv_gemm_kernel<<<(M_ROWS / 32) * (NQKV / 64) / 8, 256, 0, stream>>>(
        Xh, WhT, b_attn, Qh, Kh, VhT);

    // one wave per (b, h, q-tile): 2*16*128 = 4096 blocks of 32 threads
    lif_attention_kernel<<<BB * HH * (TT / 16), 32, 0, stream>>>(
        Qh, Kh, VhT, threshold, leak, steep, Yh);

    // (4096/32) x (1024/64) = 2048 waves, 8 waves/block -> 256 blocks
    proj_gemm_kernel<<<(M_ROWS / 32) * (CC / 64) / 8, 256, 0, stream>>>(
        Yh, WpT, b_proj, out);
}
